// MLSTM_fixD_66726611911222
// MI455X (gfx1250) — matrix operand order repaired and checked by reading, not compiled, not run on hardware
//
#include <hip/hip_runtime.h>
#include <hip/hip_bf16.h>
#include <cstdint>
#include <cstddef>

// Problem dims
#define T_   512
#define B_   256
#define IN_  128
#define H_   512
#define K_   50
#define OUT_ 10

#define NWG  32      // persistent workgroups; each owns 16 hidden units
#define NTHR 512     // 16 wave32s; wave w owns batch rows 16w..16w+15

typedef __attribute__((ext_vector_type(16))) __bf16 v16bf;
typedef __attribute__((ext_vector_type(8)))  __bf16 v8bf;
typedef __attribute__((ext_vector_type(8)))  float  v8f;
typedef __attribute__((ext_vector_type(4)))  float  v4f;

union FragBF { v16bf v; v8bf h[2]; };

__device__ inline float sigf(float x) { return 1.0f / (1.0f + __expf(-x)); }

__device__ inline v8f wmma_bf16(v16bf a, v16bf b, v8f c) {
  // D = A(16x32) * B(32x16) + C, f32 accumulate
  return __builtin_amdgcn_wmma_f32_16x16x32_bf16(false, a, false, b, (short)0, c,
                                                 false, false);
}

// A fragment (16x32 bf16, row-major source, row stride ld elements).
// lanes 0-15: row m, K 0..7 & 16..23 ; lanes 16-31: row m, K 8..15 & 24..31
__device__ inline v16bf load_fragA(const __bf16* base, int ld) {
  const int lane = threadIdx.x & 31;
  const int m  = lane & 15;
  const int kh = lane >> 4;
  const __bf16* p = base + (size_t)m * ld + kh * 8;
  FragBF f;
  f.h[0] = *(const v8bf*)(p);
  f.h[1] = *(const v8bf*)(p + 16);
  return f.v;
}

// B fragment (32x16) from LDS; weights stored row-major W[n][k] so B[k][n]=W[n][k].
// lanes 0-15: col n, K 0..7 & 16..23 ; lanes 16-31: col n, K 8..15 & 24..31
__device__ inline v16bf load_fragB_lds(const __bf16* base, int ld) {
  const int lane = threadIdx.x & 31;
  const int n  = lane & 15;
  const int kh = lane >> 4;
  const __bf16* p = base + (size_t)n * ld + kh * 8;
  FragBF f;
  f.h[0] = *(const v8bf*)(p);
  f.h[1] = *(const v8bf*)(p + 16);
  return f.v;
}

__device__ inline void grid_barrier(unsigned* bar) {
  __syncthreads();
  __threadfence();
  if (threadIdx.x == 0) {
    unsigned gen = __hip_atomic_load(bar + 1, __ATOMIC_RELAXED, __HIP_MEMORY_SCOPE_AGENT);
    unsigned prev = __hip_atomic_fetch_add(bar, 1u, __ATOMIC_ACQ_REL, __HIP_MEMORY_SCOPE_AGENT);
    if (prev == NWG - 1u) {
      __hip_atomic_store(bar, 0u, __ATOMIC_RELAXED, __HIP_MEMORY_SCOPE_AGENT);
      __hip_atomic_fetch_add(bar + 1, 1u, __ATOMIC_RELEASE, __HIP_MEMORY_SCOPE_AGENT);
    } else {
      while (__hip_atomic_load(bar + 1, __ATOMIC_ACQUIRE, __HIP_MEMORY_SCOPE_AGENT) == gen)
        __builtin_amdgcn_s_sleep(7);
    }
  }
  __syncthreads();
}

// ---------------- prep kernels ----------------
__global__ void mlstm_zero(unsigned* p, size_t nwords) {
  size_t i  = (size_t)blockIdx.x * blockDim.x + threadIdx.x;
  size_t st = (size_t)gridDim.x * blockDim.x;
  for (; i < nwords; i += st) p[i] = 0u;
}

__global__ void mlstm_cvt_bf16(const float* __restrict__ s, __bf16* __restrict__ d, size_t n) {
  size_t i  = (size_t)blockIdx.x * blockDim.x + threadIdx.x;
  size_t st = (size_t)gridDim.x * blockDim.x;
  for (; i < n; i += st) d[i] = (__bf16)s[i];
}

// wd coefficients: c_j = c_{j-1} * (j-1-d)/j, d = 0.5*sigmoid(bd); wdc[(j-1)*H + h] = c_j
__global__ void mlstm_wd(const float* __restrict__ bd, float* __restrict__ wdc) {
  int h = threadIdx.x;                       // 512 threads
  float d = 0.5f * sigf(bd[h]);
  float c = 1.0f;
  for (int j = 1; j <= K_; ++j) {
    c *= (((float)(j - 1)) - d) / (float)j;
    wdc[(size_t)(j - 1) * H_ + h] = c;
  }
}

// ---------------- persistent recurrent kernel ----------------
__global__ __launch_bounds__(NTHR, 1)
void mlstm_persist(const __bf16* __restrict__ xbf,     // (T,B,IN) bf16
                   __bf16* __restrict__ hc2,           // (B,H,64) circular c-history, bf16
                   __bf16* __restrict__ h1b,           // 2 x (B,H) bf16 (ping-pong)
                   __bf16* __restrict__ h2b,           // 2 x (B,H) bf16 (ping-pong)
                   const __bf16* __restrict__ wc,      // (3H, IN+H) bf16
                   const __bf16* __restrict__ wi,      // (4H, H) bf16
                   const __bf16* __restrict__ wh,      // (4H, H) bf16
                   const float*  __restrict__ wdc,     // (K, H) f32
                   float* __restrict__ sel,            // (B, H) f32
                   unsigned* __restrict__ bar,
                   const float* __restrict__ bcell,
                   const float* __restrict__ bih2,
                   const float* __restrict__ bhh2,
                   const int*   __restrict__ length) {
  extern __shared__ char smem[];
  constexpr int LW1 = 648;  // 640 + 8 pad (bank spread)
  constexpr int LW2 = 520;  // 512 + 8 pad
  constexpr int LWR = 68;   // rotated-wd row stride (floats): 4-bank lane spread
  __bf16* sW1 = (__bf16*)smem;            // 48 x LW1  (gates i,o,g x 16 units)
  __bf16* sWi = sW1 + 48 * LW1;           // 64 x LW2  (gates i,f,g,o x 16 units)
  __bf16* sWh = sWi + 64 * LW2;           // 64 x LW2
  float*  sC2 = (float*)(sWh + 64 * LW2); // 256 x 16 layer-2 cell state
  float*  sWd = sC2 + B_ * 16;            // 50 x 16 fractional weights (by lag j)
  float*  sWdR = sWd + K_ * 16;           // 16 x 68 per-step slot-rotated weights

  const int tid  = threadIdx.x;
  const int wg   = blockIdx.x;
  const int wv   = tid >> 5;
  const int lane = tid & 31;
  const int row0 = wv * 16;               // batch row tile owned by this wave
  const int n    = lane & 15;             // unit within WG slice
  const int hcol = wg * 16 + n;           // global hidden unit
  const int moff = (lane >> 4) * 8;       // acc VGPR r -> row M = moff + r

  // One-time staging of this WG's weight slices into LDS (resident all 512 steps).
  for (int idx = tid; idx < 48 * 320; idx += NTHR) {   // 320 dwords per 640-elem row
    int r = idx / 320, c = idx - r * 320;
    int gate = r >> 4, u = r & 15;
    *(unsigned*)(sW1 + r * LW1 + c * 2) =
      *(const unsigned*)(wc + ((size_t)(gate * H_ + wg * 16 + u)) * (IN_ + H_) + c * 2);
  }
  for (int idx = tid; idx < 64 * 256; idx += NTHR) {   // 256 dwords per 512-elem row
    int r = idx >> 8, c = idx & 255;
    int gate = r >> 4, u = r & 15;
    size_t go = ((size_t)(gate * H_ + wg * 16 + u)) * H_ + c * 2;
    *(unsigned*)(sWi + r * LW2 + c * 2) = *(const unsigned*)(wi + go);
    *(unsigned*)(sWh + r * LW2 + c * 2) = *(const unsigned*)(wh + go);
  }
  for (int idx = tid; idx < K_ * 16; idx += NTHR)
    sWd[idx] = wdc[(size_t)(idx >> 4) * H_ + wg * 16 + (idx & 15)];
  for (int idx = tid; idx < B_ * 16; idx += NTHR) sC2[idx] = 0.f;
  __syncthreads();

  const v8f vz = {0.f, 0.f, 0.f, 0.f, 0.f, 0.f, 0.f, 0.f};

  for (int t = 0; t < T_; ++t) {
    const int cur = t & 1, nxt = cur ^ 1;
    const __bf16* h1cur = h1b + (size_t)cur * B_ * H_;
    __bf16*       h1nxt = h1b + (size_t)nxt * B_ * H_;
    const __bf16* h2cur = h2b + (size_t)cur * B_ * H_;
    __bf16*       h2nxt = h2b + (size_t)nxt * B_ * H_;

    // Per-step rotated fractional weights: sWdR[u][s] = wd_j with j=(t-s)&63 (0 if j>K).
    // Slot s==t&63 gets weight 0 (j==0), so the full 64-slot row can be read blindly.
    for (int idx = tid; idx < 16 * 64; idx += NTHR) {
      int u = idx >> 6, s = idx & 63;
      int j = (t - s) & 63;
      sWdR[u * LWR + s] = (j >= 1 && j <= K_) ? sWd[(j - 1) * 16 + u] : 0.f;
    }
    __syncthreads();

    // ---------------- Stage A: gates1 = [x_t, h1] @ Wc^T ----------------
    v8f ai = vz, ao = vz, ag = vz;
    {
      const __bf16* xrow = xbf + ((size_t)t * B_ + row0) * IN_;
      if (t + 1 < T_)
        __builtin_prefetch(xbf + ((size_t)(t + 1) * B_ + row0) * IN_ + (lane << 3), 0, 3);
      for (int k0 = 0; k0 < IN_; k0 += 32) {
        v16bf a = load_fragA(xrow + k0, IN_);
        ai = wmma_bf16(a, load_fragB_lds(sW1 + 0 * 16 * LW1 + k0, LW1), ai);
        ao = wmma_bf16(a, load_fragB_lds(sW1 + 1 * 16 * LW1 + k0, LW1), ao);
        ag = wmma_bf16(a, load_fragB_lds(sW1 + 2 * 16 * LW1 + k0, LW1), ag);
      }
      const __bf16* hrow = h1cur + (size_t)row0 * H_;
      for (int k0 = 0; k0 < H_; k0 += 32) {
        v16bf a = load_fragA(hrow + k0, H_);
        ai = wmma_bf16(a, load_fragB_lds(sW1 + 0 * 16 * LW1 + IN_ + k0, LW1), ai);
        ao = wmma_bf16(a, load_fragB_lds(sW1 + 1 * 16 * LW1 + IN_ + k0, LW1), ao);
        ag = wmma_bf16(a, load_fragB_lds(sW1 + 2 * 16 * LW1 + IN_ + k0, LW1), ag);
      }
    }
    {
      // 64 rotated weights for this lane's unit, reused across all 8 row-elements.
      v4f wr[16];
#pragma unroll
      for (int q = 0; q < 16; ++q) wr[q] = *(const v4f*)(sWdR + n * LWR + q * 4);

      const float bi = bcell[hcol], bo = bcell[H_ + hcol], bg = bcell[2 * H_ + hcol];
#pragma unroll
      for (int r = 0; r < 8; ++r) {
        const int b = row0 + moff + r;
        float iv = ai[r] + bi, ov = ao[r] + bo, gv = ag[r] + bg;
        __bf16* hcr = hc2 + ((size_t)b * H_ + hcol) * 64;
        float frac = 0.f;
#pragma unroll
        for (int q = 0; q < 8; ++q) {           // 8 x 16B vector loads of history row
          v8bf hv = *(const v8bf*)(hcr + q * 8);
#pragma unroll
          for (int e = 0; e < 8; ++e)
            frac += wr[q * 2 + (e >> 2)][e & 3] * (float)hv[e];
        }
        float cv = sigf(iv) * tanhf(gv) - frac;
        hcr[t & 63] = (__bf16)cv;                         // push into circular history
        float h1n = sigf(ov) * tanhf(cv);
        h1nxt[(size_t)b * H_ + hcol] = (__bf16)h1n;
      }
    }
    grid_barrier(bar);

    // ---------------- Stage B: gates2 = h1n @ Wi^T + h2 @ Wh^T ----------------
    v8f q0 = vz, q1 = vz, q2 = vz, q3 = vz;
    {
      const __bf16* r1 = h1nxt + (size_t)row0 * H_;
      for (int k0 = 0; k0 < H_; k0 += 32) {
        v16bf a = load_fragA(r1 + k0, H_);
        q0 = wmma_bf16(a, load_fragB_lds(sWi + 0 * 16 * LW2 + k0, LW2), q0);
        q1 = wmma_bf16(a, load_fragB_lds(sWi + 1 * 16 * LW2 + k0, LW2), q1);
        q2 = wmma_bf16(a, load_fragB_lds(sWi + 2 * 16 * LW2 + k0, LW2), q2);
        q3 = wmma_bf16(a, load_fragB_lds(sWi + 3 * 16 * LW2 + k0, LW2), q3);
      }
      const __bf16* r2 = h2cur + (size_t)row0 * H_;
      for (int k0 = 0; k0 < H_; k0 += 32) {
        v16bf a = load_fragA(r2 + k0, H_);
        q0 = wmma_bf16(a, load_fragB_lds(sWh + 0 * 16 * LW2 + k0, LW2), q0);
        q1 = wmma_bf16(a, load_fragB_lds(sWh + 1 * 16 * LW2 + k0, LW2), q1);
        q2 = wmma_bf16(a, load_fragB_lds(sWh + 2 * 16 * LW2 + k0, LW2), q2);
        q3 = wmma_bf16(a, load_fragB_lds(sWh + 3 * 16 * LW2 + k0, LW2), q3);
      }
    }
    {
      const float c0 = bih2[hcol]           + bhh2[hcol];
      const float c1 = bih2[H_ + hcol]      + bhh2[H_ + hcol];
      const float c2 = bih2[2 * H_ + hcol]  + bhh2[2 * H_ + hcol];
      const float c3 = bih2[3 * H_ + hcol]  + bhh2[3 * H_ + hcol];
#pragma unroll
      for (int r = 0; r < 8; ++r) {
        const int b = row0 + moff + r;
        float i2 = q0[r] + c0, f2 = q1[r] + c1, gg = q2[r] + c2, o2 = q3[r] + c3;
        float c2v = sigf(f2) * sC2[b * 16 + n] + sigf(i2) * tanhf(gg);
        sC2[b * 16 + n] = c2v;
        float h2n = tanhf(sigf(o2) * tanhf(c2v));
        h2nxt[(size_t)b * H_ + hcol] = (__bf16)h2n;
        if (length[b] == t) sel[(size_t)b * H_ + hcol] = h2n;
      }
    }
    grid_barrier(bar);
  }
}

// ---------------- tail: logits + log_softmax ----------------
__global__ void mlstm_out(const float* __restrict__ sel, const float* __restrict__ Wout,
                          const float* __restrict__ bout, float* __restrict__ out) {
  int b = threadIdx.x;  // 256 threads
  float acc[OUT_];
#pragma unroll
  for (int o = 0; o < OUT_; ++o) acc[o] = bout[o];
  for (int k = 0; k < H_; ++k) {
    float s = sel[(size_t)b * H_ + k];
#pragma unroll
    for (int o = 0; o < OUT_; ++o) acc[o] += s * Wout[(size_t)o * H_ + k];
  }
  float mx = acc[0];
#pragma unroll
  for (int o = 1; o < OUT_; ++o) mx = fmaxf(mx, acc[o]);
  float sum = 0.f;
#pragma unroll
  for (int o = 0; o < OUT_; ++o) sum += __expf(acc[o] - mx);
  float l = __logf(sum);
#pragma unroll
  for (int o = 0; o < OUT_; ++o) out[(size_t)b * OUT_ + o] = acc[o] - mx - l;
}

// ---------------- workspace layout (bytes) ----------------
static constexpr size_t OFF_HC  = 0;                      // (B,H,64) bf16 = 16,777,216
static constexpr size_t OFF_H1  = 16777216;               // 2*(B,H) bf16 =    524,288
static constexpr size_t OFF_H2  = 17301504;               // 2*(B,H) bf16 =    524,288
static constexpr size_t OFF_BAR = 17825792;               // barrier      =        256
static constexpr size_t ZERO_WORDS = (17825792 + 256) / 4;
static constexpr size_t OFF_X   = 17826048;               // (T,B,IN) bf16 = 33,554,432
static constexpr size_t OFF_WC  = 51380480;               // (3H,640) bf16 =  1,966,080
static constexpr size_t OFF_WI  = 53346560;               // (4H,512) bf16 =  2,097,152
static constexpr size_t OFF_WH  = 55443712;               // (4H,512) bf16 =  2,097,152
static constexpr size_t OFF_WD  = 57540864;               // (K,H) f32     =    102,400
static constexpr size_t OFF_SEL = 57643264;               // (B,H) f32     =    524,288

static constexpr int SMEM_BYTES = 48 * 648 * 2 + 2 * 64 * 520 * 2 + B_ * 16 * 4 +
                                  K_ * 16 * 4 + 16 * 68 * 4;   // = 219,264

extern "C" void kernel_launch(void* const* d_in, const int* in_sizes, int n_in,
                              void* d_out, int out_size, void* d_ws, size_t ws_size,
                              hipStream_t stream) {
  (void)in_sizes; (void)n_in; (void)out_size; (void)ws_size;
  const float* inputs = (const float*)d_in[0];
  const int*   length = (const int*)d_in[1];
  const float* bd     = (const float*)d_in[2];
  const float* W_cell = (const float*)d_in[3];
  const float* b_cell = (const float*)d_in[4];
  const float* W_ih2  = (const float*)d_in[5];
  const float* W_hh2  = (const float*)d_in[6];
  const float* b_ih2  = (const float*)d_in[7];
  const float* b_hh2  = (const float*)d_in[8];
  const float* W_out  = (const float*)d_in[9];
  const float* b_out  = (const float*)d_in[10];

  char* ws = (char*)d_ws;
  __bf16* hc2 = (__bf16*)(ws + OFF_HC);
  __bf16* h1b = (__bf16*)(ws + OFF_H1);
  __bf16* h2b = (__bf16*)(ws + OFF_H2);
  unsigned* bar = (unsigned*)(ws + OFF_BAR);
  __bf16* xbf = (__bf16*)(ws + OFF_X);
  __bf16* wcb = (__bf16*)(ws + OFF_WC);
  __bf16* wib = (__bf16*)(ws + OFF_WI);
  __bf16* whb = (__bf16*)(ws + OFF_WH);
  float*  wdc = (float*)(ws + OFF_WD);
  float*  sel = (float*)(ws + OFF_SEL);

  hipFuncSetAttribute((const void*)mlstm_persist,
                      hipFuncAttributeMaxDynamicSharedMemorySize, SMEM_BYTES);

  // 1) zero state (hc history, h1/h2 ping-pong, barrier)
  mlstm_zero<<<2048, 256, 0, stream>>>((unsigned*)(ws + OFF_HC), ZERO_WORDS);
  // 2) f32 -> bf16 conversions
  mlstm_cvt_bf16<<<4096, 256, 0, stream>>>(inputs, xbf, (size_t)T_ * B_ * IN_);
  mlstm_cvt_bf16<<<1024, 256, 0, stream>>>(W_cell, wcb, (size_t)3 * H_ * (IN_ + H_));
  mlstm_cvt_bf16<<<1024, 256, 0, stream>>>(W_ih2, wib, (size_t)4 * H_ * H_);
  mlstm_cvt_bf16<<<1024, 256, 0, stream>>>(W_hh2, whb, (size_t)4 * H_ * H_);
  // 3) fractional-order coefficients
  mlstm_wd<<<1, H_, 0, stream>>>(bd, wdc);
  // 4) persistent recurrent kernel (all 512 timesteps, grid barriers between stages)
  mlstm_persist<<<NWG, NTHR, SMEM_BYTES, stream>>>(xbf, hc2, h1b, h2b, wcb, wib, whb,
                                                   wdc, sel, bar, b_cell, b_ih2, b_hh2,
                                                   length);
  // 5) output projection + log_softmax
  mlstm_out<<<1, B_, 0, stream>>>(sel, W_out, b_out, (float*)d_out);
}
